// PretrainingNetwork_37417755083163
// MI455X (gfx1250) — compile-verified, matrix operand order checked by reading, not run on hardware
//
#include <hip/hip_runtime.h>

// ---------- types ----------
typedef __attribute__((ext_vector_type(16))) __bf16 v16bf;
typedef __attribute__((ext_vector_type(8)))  float  v8f;
typedef __attribute__((ext_vector_type(4)))  unsigned int u32x4;
typedef __attribute__((ext_vector_type(8)))  int    i32x8;
typedef __attribute__((ext_vector_type(4)))  int    i32x4;

union Frag16 { v16bf v; uint4 q[2]; };

static __device__ __forceinline__ v8f wmma_bf16(v16bf a, v16bf b, v8f c) {
  return __builtin_amdgcn_wmma_f32_16x16x32_bf16(false, a, false, b, (short)0, c, false, false);
}

static __device__ __forceinline__ v8f vzero8() {
  v8f z = {0.f, 0.f, 0.f, 0.f, 0.f, 0.f, 0.f, 0.f};
  return z;
}

#define DD 1024
#define SS 2048

// ---------- Tensor Data Mover: 2D tile Global -> LDS ----------
// D# packing per CDNA5 ISA:
//  group0: [1:0]=count(1), [63:32]=lds_addr, [120:64]=global_addr, [127:126]=type(2)
//  group1: [17:16]=data_size(1 -> 2B), [79:48]=tensor_dim0, [111:80]=tensor_dim1,
//          [127:112]=tile_dim0, [143:128]=tile_dim1, [207:160]=tensor_dim0_stride
//  groups 2/3 zero (<=2D tensor). Dims/strides in data_size units.
#if __has_include(<hip/amd_detail/amd_gfx1250_TDM.h>)
#define TDM_6ARG 1
#else
#define TDM_6ARG 0
#endif

static __device__ __forceinline__ void tdm_load_2d_bf16(
    unsigned lds_addr, const void* gp, unsigned tile_d0, unsigned tile_d1,
    unsigned tensor_d0, unsigned tensor_d1, unsigned stride0) {
  unsigned long long ga = (unsigned long long)(uintptr_t)gp;
  u32x4 g0 = { 1u, lds_addr, (unsigned)ga,
               ((unsigned)((ga >> 32) & 0x01FFFFFFull)) | 0x80000000u };
  i32x8 g1 = {
      (int)(1u << 16),                                            // data_size = 2B
      (int)((tensor_d0 & 0xFFFFu) << 16),                         // tensor_dim0 lo
      (int)((tensor_d0 >> 16) | ((tensor_d1 & 0xFFFFu) << 16)),   // dim0 hi | dim1 lo
      (int)((tensor_d1 >> 16) | ((tile_d0 & 0xFFFFu) << 16)),     // dim1 hi | tile0
      (int)(tile_d1 & 0xFFFFu),                                   // tile1 | tile2(0)
      (int)stride0,                                               // dim0_stride lo
      0, 0 };
  i32x4 z4 = {0, 0, 0, 0};
#if TDM_6ARG
  i32x8 z8 = {0, 0, 0, 0, 0, 0, 0, 0};
  __builtin_amdgcn_tensor_load_to_lds(g0, g1, z4, z4, z8, 0);
#else
  __builtin_amdgcn_tensor_load_to_lds(g0, g1, z4, z4, 0);
#endif
}

// ---------- weight f32 -> bf16 ----------
__global__ void cvt_bf16(const float* __restrict__ s, __bf16* __restrict__ d, int n) {
  int i = blockIdx.x * 256 + threadIdx.x;
  if (i < n) d[i] = (__bf16)s[i];
}

// ---------- generic WMMA GEMM: out(bf16) = act(A[16384xD] @ W[DxD] + bias [+resid]) ----------
// Tile: 128 rows x 128 cols, 8 waves (256 thr), K-step 32.
// Wave w owns m-subtile w (16 rows) and all 8 n-subtiles -> 8 v8f accumulators.
// B tiles staged in LDS in fragment-swizzled order: element (k,n) at
//   [n>>4][ (n&15) | ((k&8)<<1) ][ ((k&16)>>1) | (k&7) ]
// so a B fragment is one contiguous 32B read per lane.
// bf16-A path: A tile staged by the Tensor Data Mover (wave 0 issues, TENSORcnt wait).
template <typename AT, bool TRANS_OUT, bool RESID, bool RELU>
__global__ __launch_bounds__(256) void gemm_wmma(
    const AT* __restrict__ A, const __bf16* __restrict__ W,
    const float* __restrict__ bias, const float* __restrict__ resid,
    __bf16* __restrict__ out) {
  __shared__ __align__(16) __bf16 As[128 * 32];
  __shared__ __align__(16) __bf16 Bs[8 * 32 * 16];

  const int row_base = blockIdx.x * 128;
  const int n_base = blockIdx.y * 128;
  const int t = threadIdx.x, lane = t & 31, wave = t >> 5;
  const int hi = lane >> 4, nl = lane & 15;

  v8f acc[8];
#pragma unroll
  for (int i = 0; i < 8; ++i) acc[i] = vzero8();

  for (int kt = 0; kt < DD; kt += 32) {
    __syncthreads();
    // ---- stage A (128x32) ----
    if constexpr (sizeof(AT) == 2) {
      // TDM: strided 2D tile (128 rows x 32 cols, row stride DD), packed into As.
      if (t < 32) {
        tdm_load_2d_bf16((unsigned)(uintptr_t)As,
                         A + (size_t)row_base * DD + kt,
                         /*tile_d0=*/32, /*tile_d1=*/128,
                         /*tensor_d0=*/32, /*tensor_d1=*/128, /*stride0=*/DD);
      }
    } else {
      const int r = t >> 1, half = t & 1;
      const float* sf = (const float*)(A + (size_t)(row_base + r) * DD + kt + half * 16);
      __bf16* dst = As + r * 32 + half * 16;
      __builtin_prefetch((const void*)(sf + 32), 0, 0);
#pragma unroll
      for (int i = 0; i < 16; i += 4) {
        float4 f = *(const float4*)(sf + i);
        dst[i + 0] = (__bf16)f.x; dst[i + 1] = (__bf16)f.y;
        dst[i + 2] = (__bf16)f.z; dst[i + 3] = (__bf16)f.w;
      }
    }
    // ---- stage W chunk (32x128) swizzled ----
    {
      const int k = t >> 3, n0 = (t & 7) * 16;
      const __bf16* src = W + (size_t)(kt + k) * DD + n_base + n0;
      __builtin_prefetch((const void*)(src + 32 * DD), 0, 0);
      const int e = ((k & 16) >> 1) | (k & 7);
      const int lb = (k & 8) << 1;
#pragma unroll
      for (int i = 0; i < 16; ++i) {
        int n = n0 + i;
        Bs[(n >> 4) * 512 + ((n & 15) | lb) * 16 + e] = src[i];
      }
    }
    if constexpr (sizeof(AT) == 2) {
      if (t < 32) __builtin_amdgcn_s_wait_tensorcnt(0);
    }
    __syncthreads();
    // ---- compute: 8 WMMAs ----
    Frag16 fa, fb;
    const __bf16* ap = As + (wave * 16 + nl) * 32 + 8 * hi;
    fa.q[0] = *(const uint4*)ap;
    fa.q[1] = *(const uint4*)(ap + 16);
#pragma unroll
    for (int nt = 0; nt < 8; ++nt) {
      const __bf16* bp = Bs + nt * 512 + lane * 16;
      fb.q[0] = ((const uint4*)bp)[0];
      fb.q[1] = ((const uint4*)bp)[1];
      acc[nt] = wmma_bf16(fa.v, fb.v, acc[nt]);
    }
  }

  // ---- epilogue ----
#pragma unroll
  for (int nt = 0; nt < 8; ++nt) {
    const int n = n_base + nt * 16 + nl;
    const float bv = bias[n];
#pragma unroll
    for (int r = 0; r < 8; ++r) {
      const int grow = row_base + wave * 16 + r + 8 * hi;
      float vv = acc[nt][r] + bv;
      if constexpr (RESID) vv += resid[(size_t)grow * DD + n];
      if constexpr (RELU) vv = vv > 0.f ? vv : 0.f;
      __bf16 ov = (__bf16)vv;
      if constexpr (TRANS_OUT) {
        const int b = grow >> 11, s = grow & (SS - 1);
        out[((size_t)b * DD + n) * SS + s] = ov;  // K stored [b][d][s]
      } else {
        out[(size_t)grow * DD + n] = ov;
      }
    }
  }
}

// ---------- flash attention: att = softmax(Q K^T) V, bf16 in/out ----------
// Grid: (S/16, B). 8 waves. Wave w: score n-subtile w of the 16x128 tile,
// and output d-columns [128w, 128w+128). Q tile staged by the TDM.
__global__ __launch_bounds__(256) void flash_attn(
    const __bf16* __restrict__ qg, const __bf16* __restrict__ ktg,
    const __bf16* __restrict__ vg, __bf16* __restrict__ att) {
  extern __shared__ __align__(16) char smem[];
  __bf16* Qt   = (__bf16*)(smem);            // 16x1024        (32768 B)
  __bf16* KtSw = (__bf16*)(smem + 32768);    // 2 ksteps x 8 nsub x 32 x 16 (16384 B)
  float*  Sc   = (float*) (smem + 49152);    // 16x128         (8192 B)
  __bf16* P    = (__bf16*)(smem + 57344);    // 16x128         (4096 B)
  __bf16* VtSw = (__bf16*)(smem + 61440);    // 64 nsub x 32 x 16 (65536 B)
  float*  mrow = (float*) (smem + 126976);   // 16
  float*  lrow = mrow + 16;
  float*  srow = mrow + 32;

  const int b = blockIdx.y, q0 = blockIdx.x * 16;
  const int t = threadIdx.x, lane = t & 31, wave = t >> 5;
  const int hi = lane >> 4, nl = lane & 15;

  // load Q tile (16x1024 bf16, contiguous) via Tensor Data Mover
  if (t < 32) {
    tdm_load_2d_bf16((unsigned)(uintptr_t)Qt,
                     qg + ((size_t)b * SS + q0) * DD,
                     /*tile_d0=*/DD, /*tile_d1=*/16,
                     /*tensor_d0=*/DD, /*tensor_d1=*/16, /*stride0=*/DD);
    __builtin_amdgcn_s_wait_tensorcnt(0);
  }
  if (t < 16) { mrow[t] = -3.0e38f; lrow[t] = 0.f; }

  v8f o[8];
#pragma unroll
  for (int i = 0; i < 8; ++i) o[i] = vzero8();
  __syncthreads();

  for (int kv = 0; kv < SS; kv += 128) {
    // ---- scores: S_tile[16][128] = Q @ K^T, streamed over D ----
    v8f sacc = vzero8();
    for (int dc = 0; dc < DD; dc += 64) {
      __syncthreads();
      // stage K^T chunk: k-dim = d (64), n = kv (128), swizzled per 32-k step
      for (int i = 0; i < 32; ++i) {
        int lin = i * 256 + t;            // 0..8191
        int k = lin >> 7, n = lin & 127;
        const __bf16* kp = ktg + ((size_t)b * DD + dc + k) * SS + kv + n;
        __builtin_prefetch((const void*)(kp + 64 * SS), 0, 0);
        __bf16 val = *kp;
        int e = ((k & 16) >> 1) | (k & 7);
        KtSw[(k >> 5) * 4096 + (n >> 4) * 512 + (((n & 15) | ((k & 8) << 1))) * 16 + e] = val;
      }
      __syncthreads();
      Frag16 fa, fb;
#pragma unroll
      for (int ks = 0; ks < 2; ++ks) {
        const __bf16* ap = Qt + nl * DD + dc + ks * 32 + 8 * hi;
        fa.q[0] = *(const uint4*)ap;
        fa.q[1] = *(const uint4*)(ap + 16);
        const __bf16* bp = KtSw + ks * 4096 + wave * 512 + lane * 16;
        fb.q[0] = ((const uint4*)bp)[0];
        fb.q[1] = ((const uint4*)bp)[1];
        sacc = wmma_bf16(fa.v, fb.v, sacc);
      }
    }
    // ---- spill scores to LDS ----
#pragma unroll
    for (int r = 0; r < 8; ++r)
      Sc[(r + 8 * hi) * 128 + wave * 16 + nl] = sacc[r];
    __syncthreads();
    // ---- online softmax (16 threads, one row each) ----
    if (t < 16) {
      float mo = mrow[t], mx = mo;
      for (int c = 0; c < 128; ++c) mx = fmaxf(mx, Sc[t * 128 + c]);
      float sc = __expf(mo - mx), sum = 0.f;
      for (int c = 0; c < 128; ++c) {
        float p = __expf(Sc[t * 128 + c] - mx);
        P[t * 128 + c] = (__bf16)p;
        sum += p;
      }
      lrow[t] = lrow[t] * sc + sum;
      mrow[t] = mx;
      srow[t] = sc;
    }
    __syncthreads();
    // ---- rescale running output ----
    float sc8[8];
#pragma unroll
    for (int r = 0; r < 8; ++r) sc8[r] = srow[r + 8 * hi];
#pragma unroll
    for (int nt = 0; nt < 8; ++nt)
#pragma unroll
      for (int r = 0; r < 8; ++r) o[nt][r] = o[nt][r] * sc8[r];
    // ---- P @ V over 4 k-steps of 32 kv rows ----
    for (int kvs = 0; kvs < 4; ++kvs) {
      __syncthreads();
      // stage V chunk 32(kv) x 1024(d), swizzled
      for (int i = 0; i < 16; ++i) {
        int lin4 = i * 256 + t;            // uint4 index, 4096 total
        int elem = lin4 * 8;
        int k = elem >> 10, n0 = elem & (DD - 1);
        uint4 raw = *(const uint4*)(vg + ((size_t)b * SS + kv + kvs * 32 + k) * DD + n0);
        const __bf16* pv = (const __bf16*)&raw;
        int e = ((k & 16) >> 1) | (k & 7);
        int lb = (k & 8) << 1;
#pragma unroll
        for (int j = 0; j < 8; ++j) {
          int n = n0 + j;
          VtSw[(n >> 4) * 512 + ((n & 15) | lb) * 16 + e] = pv[j];
        }
      }
      __syncthreads();
      Frag16 fa, fb;
      const __bf16* ap = P + nl * 128 + kvs * 32 + 8 * hi;
      fa.q[0] = *(const uint4*)ap;
      fa.q[1] = *(const uint4*)(ap + 16);
#pragma unroll
      for (int nt = 0; nt < 8; ++nt) {
        const __bf16* bp = VtSw + (wave * 8 + nt) * 512 + lane * 16;
        fb.q[0] = ((const uint4*)bp)[0];
        fb.q[1] = ((const uint4*)bp)[1];
        o[nt] = wmma_bf16(fa.v, fb.v, o[nt]);
      }
    }
  }
  __syncthreads();
  // ---- finalize: divide by l, store bf16 ----
  float li[8];
#pragma unroll
  for (int r = 0; r < 8; ++r) li[r] = 1.0f / lrow[r + 8 * hi];
#pragma unroll
  for (int nt = 0; nt < 8; ++nt) {
    const int col = wave * 128 + nt * 16 + nl;
#pragma unroll
    for (int r = 0; r < 8; ++r) {
      const int grow = q0 + r + 8 * hi;
      att[((size_t)b * SS + grow) * DD + col] = (__bf16)(o[nt][r] * li[r]);
    }
  }
}

// ---------- final projection: out[row] = x[row,:] . W2 + b2 ----------
__global__ __launch_bounds__(256) void rowdot(
    const __bf16* __restrict__ x, const float* __restrict__ w2,
    const float* __restrict__ b2, float* __restrict__ out) {
  const int row = blockIdx.x * 8 + (threadIdx.x >> 5);
  const int lane = threadIdx.x & 31;
  const __bf16* xr = x + (size_t)row * DD;
  float acc = 0.f;
  for (int i = lane; i < DD; i += 32) acc += (float)xr[i] * w2[i];
#pragma unroll
  for (int off = 16; off > 0; off >>= 1) acc += __shfl_down(acc, off, 32);
  if (lane == 0) out[row] = acc + b2[0];
}

// ---------- launch ----------
extern "C" void kernel_launch(void* const* d_in, const int* in_sizes, int n_in,
                              void* d_out, int out_size, void* d_ws, size_t ws_size,
                              hipStream_t stream) {
  const float* visual = (const float*)d_in[0];
  const float* audio  = (const float*)d_in[1];
  const float* Wq = (const float*)d_in[2];  const float* bq = (const float*)d_in[3];
  const float* Wk = (const float*)d_in[4];  const float* bk = (const float*)d_in[5];
  const float* Wv = (const float*)d_in[6];  const float* bv = (const float*)d_in[7];
  const float* Wo = (const float*)d_in[8];  const float* bo = (const float*)d_in[9];
  const float* W1 = (const float*)d_in[10]; const float* b1 = (const float*)d_in[11];
  const float* W2 = (const float*)d_in[12]; const float* b2 = (const float*)d_in[13];

  char* ws = (char*)d_ws;
  const size_t MB = 1ull << 20;
  __bf16* qb  = (__bf16*)(ws);              // 32 MB; reused as x later
  __bf16* ktb = (__bf16*)(ws + 32 * MB);    // 32 MB, [b][d][s]
  __bf16* vb  = (__bf16*)(ws + 64 * MB);    // 32 MB
  __bf16* atb = (__bf16*)(ws + 96 * MB);    // 32 MB
  __bf16* hb  = (__bf16*)(ws + 128 * MB);   // 32 MB
  __bf16* Wqb = (__bf16*)(ws + 160 * MB);
  __bf16* Wkb = (__bf16*)(ws + 162 * MB);
  __bf16* Wvb = (__bf16*)(ws + 164 * MB);
  __bf16* Wob = (__bf16*)(ws + 166 * MB);
  __bf16* W1b = (__bf16*)(ws + 168 * MB);

  const int WN = 1 << 20;
  cvt_bf16<<<WN / 256, 256, 0, stream>>>(Wq, Wqb, WN);
  cvt_bf16<<<WN / 256, 256, 0, stream>>>(Wk, Wkb, WN);
  cvt_bf16<<<WN / 256, 256, 0, stream>>>(Wv, Wvb, WN);
  cvt_bf16<<<WN / 256, 256, 0, stream>>>(Wo, Wob, WN);
  cvt_bf16<<<WN / 256, 256, 0, stream>>>(W1, W1b, WN);

  dim3 gg(128, 8);  // 16384/128 row tiles x 1024/128 col tiles
  gemm_wmma<float, false, false, false><<<gg, 256, 0, stream>>>(visual, Wqb, bq, nullptr, qb);
  gemm_wmma<float, true,  false, false><<<gg, 256, 0, stream>>>(audio,  Wkb, bk, nullptr, ktb);
  gemm_wmma<float, false, false, false><<<gg, 256, 0, stream>>>(audio,  Wvb, bv, nullptr, vb);

  flash_attn<<<dim3(SS / 16, 8), 256, 127168, stream>>>(qb, ktb, vb, atb);

  gemm_wmma<__bf16, false, true,  false><<<gg, 256, 0, stream>>>(atb, Wob, bo, visual, hb);
  gemm_wmma<__bf16, false, false, true ><<<gg, 256, 0, stream>>>(hb,  W1b, b1, nullptr, qb);

  rowdot<<<16384 / 8, 256, 0, stream>>>(qb, W2, b2, (float*)d_out);
}
